// Net_41721312313942
// MI455X (gfx1250) — matrix operand order repaired and checked by reading, not compiled, not run on hardware
//
#include <hip/hip_runtime.h>

typedef __attribute__((ext_vector_type(16))) _Float16 v16h;
typedef __attribute__((ext_vector_type(8)))  _Float16 v8h;
typedef __attribute__((ext_vector_type(8)))  float    v8f;

#define BATCH   65536
#define HID     50
#define NLAY    6
#define INSZ    3
#define SEQT    3
#define KP      64      // padded K dimension (input or hidden)
#define NP      256     // 4 gates * 64 padded hidden
#define ROWS    64      // batch rows per block
#define THREADS 128     // 4 waves * 32 lanes

// ---------------- fast activations on the gfx1250 transcendental unit ----------------
__device__ __forceinline__ float fast_tanh(float x) {
#if __has_builtin(__builtin_amdgcn_tanhf)
    return __builtin_amdgcn_tanhf(x);            // v_tanh_f32
#elif __has_builtin(__builtin_amdgcn_tanh_f32)
    return __builtin_amdgcn_tanh_f32(x);
#else
    float e = __expf(2.0f * x);
    return 1.0f - 2.0f / (e + 1.0f);
#endif
}
__device__ __forceinline__ float sigm(float x) {
    return 0.5f * fast_tanh(0.5f * x) + 0.5f;
}

// LDS byte offset of a __shared__ pointer (generic -> addrspace(3) -> offset)
__device__ __forceinline__ unsigned ldsOff(const void* p) {
    return (unsigned)(unsigned long long)(__attribute__((address_space(3))) const void*)p;
}

// A operand: 16x32 f16 tile from row-major [row][KP] LDS.
// lanes 0-15 (M=lane): K {kt*32+0..7, kt*32+16..23}; lanes 16-31 (M=lane-16): K {8..15, 24..31}.
__device__ __forceinline__ v16h load_A(const _Float16* rowBasePtr, int lane, int ktile) {
    int m    = lane & 15;
    int half = lane >> 4;
    const _Float16* p = rowBasePtr + m * KP + ktile * 32 + half * 8;
    v8h lo = *(const v8h*)(p);
    v8h hi = *(const v8h*)(p + 16);
    return __builtin_shufflevector(lo, hi, 0,1,2,3,4,5,6,7,8,9,10,11,12,13,14,15);
}

// B operand: 32x16 f16 tile (KxN) from weight LDS stored [N=256][K=64] row-major.
// lane l<16: column N=nt*16+l, K kt*32+{0..7,16..23}; lane>=16: N=nt*16+l-16, K kt*32+{8..15,24..31}.
__device__ __forceinline__ v16h load_B(const _Float16* w, int lane, int ntile, int ktile) {
    int n  = ntile * 16 + (lane & 15);
    int kb = ktile * 32 + ((lane >> 4) ? 8 : 0);
    const _Float16* p = w + n * KP + kb;
    v8h lo = *(const v8h*)(p);
    v8h hi = *(const v8h*)(p + 16);
    return __builtin_shufflevector(lo, hi, 0,1,2,3,4,5,6,7,8,9,10,11,12,13,14,15);
}

// ---------------- prep kernel: one-time f32 -> f16 weight reorganization ----------------
// Layout per layer: [gate*64 + j][k] with j,k zero-padded; gate order i,f,g,o.
__global__ __launch_bounds__(256)
void prep_weights(const float* __restrict__ w_ih0, const float* __restrict__ w_ih,
                  const float* __restrict__ w_hh,  const float* __restrict__ b_ih,
                  const float* __restrict__ b_hh,  const float* __restrict__ fc_w,
                  _Float16* __restrict__ gWih, _Float16* __restrict__ gWhh,
                  float* __restrict__ gBias, float* __restrict__ gFcw)
{
    int idx = blockIdx.x * blockDim.x + threadIdx.x;
    const int total = NLAY * NP * KP;
    if (idx < total) {
        int k    = idx & (KP - 1);
        int np   = (idx >> 6) & (NP - 1);
        int l    = idx / (NP * KP);
        int gate = np >> 6;
        int j    = np & 63;
        const float* wih_src = (l == 0) ? w_ih0 : (w_ih + (size_t)(l - 1) * 4 * HID * HID);
        const int    kin     = (l == 0) ? INSZ : HID;
        const float* whh_src = w_hh + (size_t)l * 4 * HID * HID;
        float vi = 0.0f, vh = 0.0f;
        if (j < HID) {
            if (k < kin) vi = wih_src[(size_t)(gate * HID + j) * kin + k];
            if (k < HID) vh = whh_src[(size_t)(gate * HID + j) * HID + k];
        }
        gWih[idx] = (_Float16)vi;
        gWhh[idx] = (_Float16)vh;
    }
    if (idx < NLAY * NP) {
        int l = idx >> 8, np = idx & 255;
        int gate = np >> 6, j = np & 63;
        gBias[idx] = (j < HID)
                   ? (b_ih[l * 4 * HID + gate * HID + j] + b_hh[l * 4 * HID + gate * HID + j])
                   : 0.0f;
    }
    if (idx < KP) gFcw[idx] = (idx < HID) ? fc_w[idx] : 0.0f;
}

// ---------------- main LSTM kernel ----------------
__global__ __launch_bounds__(THREADS)
void lstm_wmma_kernel(const float* __restrict__ x,
                      const float* __restrict__ h0,
                      const float* __restrict__ c0,
                      const _Float16* __restrict__ gWih,
                      const _Float16* __restrict__ gWhh,
                      const float* __restrict__ gBias,
                      const float* __restrict__ gFcw,
                      const float* __restrict__ fc_b,
                      float* __restrict__ out)
{
    __shared__ _Float16 sWih[NP * KP];              // 32 KB
    __shared__ _Float16 sWhh[NP * KP];              // 32 KB
    __shared__ float    sBias[NP];                  // 1 KB
    __shared__ float    sFcw[KP];                   // 256 B
    __shared__ _Float16 sYsA[SEQT * ROWS * KP];     // 24 KB  [t][row][k]
    __shared__ _Float16 sYsB[SEQT * ROWS * KP];     // 24 KB
    __shared__ _Float16 sHini[ROWS * KP];           // 8 KB

    const int tid   = threadIdx.x;
    const int lane  = tid & 31;
    const int wave  = tid >> 5;
    const int rowT  = wave * 16;
    const int bBase = blockIdx.x * ROWS;

    for (int i = tid; i < KP; i += THREADS) sFcw[i] = gFcw[i];

    // stage layer-0 input x[b, i, t] -> sYsA[t][row][k]
    for (int i = tid; i < SEQT * ROWS * KP; i += THREADS) {
        int k   = i & (KP - 1);
        int row = (i >> 6) & (ROWS - 1);
        int t   = i / (ROWS * KP);
        float v = 0.0f;
        if (k < INSZ) v = x[(size_t)(bBase + row) * INSZ * SEQT + k * SEQT + t];
        sYsA[i] = (_Float16)v;
    }

    _Float16* ysIn  = sYsA;
    _Float16* ysOut = sYsB;

    for (int l = 0; l < NLAY; ++l) {
        __syncthreads();  // previous layer fully consumed before restaging weights

        // ---- async DMA the pre-converted f16 weights into LDS (ASYNCcnt path) ----
        {
            const _Float16* srcIh = gWih + (size_t)l * NP * KP;
            const _Float16* srcHh = gWhh + (size_t)l * NP * KP;
            for (int i = tid * 8; i < NP * KP; i += THREADS * 8) {   // 16B per lane per op
                unsigned dIh = ldsOff(&sWih[i]);
                unsigned dHh = ldsOff(&sWhh[i]);
                int gof = i * (int)sizeof(_Float16);
                asm volatile("global_load_async_to_lds_b128 %0, %1, %2"
                             :: "v"(dIh), "v"(gof), "s"(srcIh) : "memory");
                asm volatile("global_load_async_to_lds_b128 %0, %1, %2"
                             :: "v"(dHh), "v"(gof), "s"(srcHh) : "memory");
            }
        }
        for (int i = tid; i < NP; i += THREADS) sBias[i] = gBias[l * NP + i];

        // ---- stage h0 for this layer (f32 -> f16, padded) ----
        for (int i = tid; i < ROWS * KP; i += THREADS) {
            int k   = i & (KP - 1);
            int row = i >> 6;
            float v = (k < HID) ? h0[((size_t)l * BATCH + bBase + row) * HID + k] : 0.0f;
            sHini[i] = (_Float16)v;
        }

        // ---- prefetch next layer's h0/c0 HBM streams while this layer computes ----
        if (l + 1 < NLAY) {
            const float* nh = h0 + ((size_t)(l + 1) * BATCH + bBase) * HID;
            const float* nc = c0 + ((size_t)(l + 1) * BATCH + bBase) * HID;
            for (int i = tid; i < 100; i += THREADS) {    // 64*50*4B = 12.8KB = 100 lines
                __builtin_prefetch(nh + i * 32, 0, 0);
                __builtin_prefetch(nc + i * 32, 0, 0);
            }
        }

        asm volatile("s_wait_asynccnt 0x0" ::: "memory");
        __syncthreads();

        // ---- load c0 into accumulator (C/D) register layout ----
        float cst[4][8];
        #pragma unroll
        for (int ht = 0; ht < 4; ++ht) {
            #pragma unroll
            for (int r = 0; r < 8; ++r) {
                int row = rowT + r + ((lane >> 4) ? 8 : 0);
                int col = ht * 16 + (lane & 15);
                cst[ht][r] = (col < HID)
                           ? c0[((size_t)l * BATCH + bBase + row) * HID + col]
                           : 0.0f;
            }
        }

        for (int t = 0; t < SEQT; ++t) {
            const _Float16* Ax = ysIn + ((size_t)t * ROWS + rowT) * KP;
            const _Float16* Ah = (t == 0) ? (sHini + (size_t)rowT * KP)
                                          : (ysOut + ((size_t)(t - 1) * ROWS + rowT) * KP);
            v16h ax0 = load_A(Ax, lane, 0);
            v16h ax1 = load_A(Ax, lane, 1);
            v16h ah0 = load_A(Ah, lane, 0);
            v16h ah1 = load_A(Ah, lane, 1);

            _Float16* hdst = ysOut + (size_t)t * ROWS * KP;

            #pragma unroll
            for (int ht = 0; ht < 4; ++ht) {
                v8f acc[4];
                #pragma unroll
                for (int g = 0; g < 4; ++g) {
                    float bv = sBias[g * 64 + ht * 16 + (lane & 15)];
                    v8f a;
                    #pragma unroll
                    for (int r = 0; r < 8; ++r) a[r] = bv;
                    const int nt = g * 4 + ht;
                    a = __builtin_amdgcn_wmma_f32_16x16x32_f16(
                            false, ax0, false, load_B(sWih, lane, nt, 0), (short)0, a, false, false);
                    a = __builtin_amdgcn_wmma_f32_16x16x32_f16(
                            false, ax1, false, load_B(sWih, lane, nt, 1), (short)0, a, false, false);
                    a = __builtin_amdgcn_wmma_f32_16x16x32_f16(
                            false, ah0, false, load_B(sWhh, lane, nt, 0), (short)0, a, false, false);
                    a = __builtin_amdgcn_wmma_f32_16x16x32_f16(
                            false, ah1, false, load_B(sWhh, lane, nt, 1), (short)0, a, false, false);
                    acc[g] = a;
                }
                // elementwise LSTM cell update on the C/D layout
                #pragma unroll
                for (int r = 0; r < 8; ++r) {
                    float iv = sigm(acc[0][r]);
                    float fv = sigm(acc[1][r]);
                    float gv = fast_tanh(acc[2][r]);
                    float ov = sigm(acc[3][r]);
                    float cn = fv * cst[ht][r] + iv * gv;
                    cst[ht][r] = cn;
                    float hv = ov * fast_tanh(cn);
                    int row = rowT + r + ((lane >> 4) ? 8 : 0);
                    int col = ht * 16 + (lane & 15);
                    hdst[row * KP + col] = (_Float16)hv;
                }
            }
            __syncthreads();  // cross-lane h dependence through LDS before next timestep
        }

        _Float16* tmp = ysIn; ysIn = ysOut; ysOut = tmp;
    }

    // ---- final FC: out[b] = h_last . fc_w + fc_b ----
    __syncthreads();
    const _Float16* hf = ysIn + (size_t)(SEQT - 1) * ROWS * KP;
    if (tid < ROWS) {
        float acc = fc_b[0];
        #pragma unroll
        for (int k = 0; k < HID; ++k)
            acc += (float)hf[tid * KP + k] * sFcw[k];
        out[bBase + tid] = acc;
    }
}

extern "C" void kernel_launch(void* const* d_in, const int* in_sizes, int n_in,
                              void* d_out, int out_size, void* d_ws, size_t ws_size,
                              hipStream_t stream) {
    const float* x     = (const float*)d_in[0];
    const float* h0    = (const float*)d_in[1];
    const float* c0    = (const float*)d_in[2];
    const float* w_ih0 = (const float*)d_in[3];
    const float* w_ih  = (const float*)d_in[4];
    const float* w_hh  = (const float*)d_in[5];
    const float* b_ih  = (const float*)d_in[6];
    const float* b_hh  = (const float*)d_in[7];
    const float* fc_w  = (const float*)d_in[8];
    const float* fc_b  = (const float*)d_in[9];
    float* out = (float*)d_out;

    // workspace layout: f16 weight tables + f32 bias/fc tables (~400 KB)
    _Float16* gWih = (_Float16*)d_ws;
    _Float16* gWhh = gWih + (size_t)NLAY * NP * KP;
    float*    gBias = (float*)(gWhh + (size_t)NLAY * NP * KP);
    float*    gFcw  = gBias + NLAY * NP;

    {
        int total = NLAY * NP * KP;
        dim3 grid((total + 255) / 256), block(256);
        hipLaunchKernelGGL(prep_weights, grid, block, 0, stream,
                           w_ih0, w_ih, w_hh, b_ih, b_hh, fc_w,
                           gWih, gWhh, gBias, gFcw);
    }
    {
        dim3 grid(BATCH / ROWS), block(THREADS);
        hipLaunchKernelGGL(lstm_wmma_kernel, grid, block, 0, stream,
                           x, h0, c0, gWih, gWhh, gBias, gFcw, fc_b, out);
    }
}
